// GAT_MUTAG_66116726554991
// MI455X (gfx1250) — matrix operand order repaired and checked by reading, not compiled, not run on hardware
//
#include <hip/hip_runtime.h>
#include <hip/hip_bf16.h>
#include <math.h>

typedef __attribute__((ext_vector_type(16))) _Float16 v16h;
typedef __attribute__((ext_vector_type(8)))  float    v8f;

#define HID 32
#define NEG_SLOPE 0.2f
#define BN_EPS 1e-5f

// ---------------------------------------------------------------------------
// helpers
// ---------------------------------------------------------------------------
__device__ __forceinline__ void atomicMaxFloat(float* addr, float v) {
  // classic monotone-bit trick: works for all finite floats
  if (v >= 0.0f) atomicMax((int*)addr, __float_as_int(v));
  else           atomicMin((unsigned int*)addr, __float_as_uint(v));
}

// ---------------------------------------------------------------------------
// zero helper
// ---------------------------------------------------------------------------
__global__ void zero_f32(float* __restrict__ p, int n) {
  int i = blockIdx.x * blockDim.x + threadIdx.x;
  if (i < n) p[i] = 0.0f;
}

// ---------------------------------------------------------------------------
// pad x [N,7] f32 -> [N,32] f16 (zero padded)
// ---------------------------------------------------------------------------
__global__ void prep_x(const float* __restrict__ x, _Float16* __restrict__ x16,
                       int N, int K) {
  int n = blockIdx.x * blockDim.x + threadIdx.x;
  if (n >= N) return;
  const float* xr = x + (size_t)n * K;
  _Float16* o = x16 + (size_t)n * HID;
#pragma unroll
  for (int c = 0; c < HID; ++c)
    o[c] = (c < K) ? (_Float16)xr[c] : (_Float16)0.0f;
}

// ---------------------------------------------------------------------------
// pad W [K,32] f32 -> [32,32] f16 (rows >= K zero)
// ---------------------------------------------------------------------------
__global__ void prep_W(const float* __restrict__ W, int K,
                       _Float16* __restrict__ Wp) {
  int t = threadIdx.x;           // 256 threads, 4 elems each
#pragma unroll
  for (int j = 0; j < 4; ++j) {
    int i = t * 4 + j;
    int r = i >> 5, c = i & 31;
    Wp[i] = (r < K) ? (_Float16)W[r * HID + c] : (_Float16)0.0f;
  }
}

// ---------------------------------------------------------------------------
// per-layer init: zero BN accumulators, fold edge projection:
//   c4[k] = sum_j We[k][j] * a_edge[j]   (so (ea@We)·a_edge == ea·c4)
// ---------------------------------------------------------------------------
__global__ void init_layer(const float* __restrict__ We,      // [4,32]
                           const float* __restrict__ a_edge,  // [32]
                           float* __restrict__ c4,            // [4]
                           float* __restrict__ bnsum,         // [32]
                           float* __restrict__ bnsq) {        // [32]
  int t = threadIdx.x;
  if (t < HID) { bnsum[t] = 0.0f; bnsq[t] = 0.0f; }
  if (t < 4) {
    float s = 0.0f;
#pragma unroll
    for (int j = 0; j < HID; ++j) s += We[t * HID + j] * a_edge[j];
    c4[t] = s;
  }
}

// ---------------------------------------------------------------------------
// WMMA GEMM:  H[N,32] = A16[N,32] @ B16[32,32]   (f16 in, f32 out)
// one wave per 16-row tile; two 16x16 column tiles per wave
// ---------------------------------------------------------------------------
__global__ __launch_bounds__(256)
void gemm_h(const _Float16* __restrict__ A, const _Float16* __restrict__ Bg,
            float* __restrict__ H, int nRowTiles) {
  __shared__ _Float16 Bs[HID * HID];        // 2 KB
  int tid = threadIdx.x;
#pragma unroll
  for (int i = 0; i < 4; ++i) Bs[tid * 4 + i] = Bg[tid * 4 + i];
  __syncthreads();

  int wave = tid >> 5, lane = tid & 31;
  int rowTile = blockIdx.x * 8 + wave;
  if (rowTile >= nRowTiles) return;        // wave-uniform -> EXEC all-ones below
  int row0 = rowTile << 4;

  // ---- A fragment: 16-bit A 16x32 layout ----
  int mrow = lane & 15;
  int kb   = (lane < 16) ? 0 : 8;
  const _Float16* ar = A + (size_t)(row0 + mrow) * HID;
  v16h a;
#pragma unroll
  for (int i = 0; i < 8; ++i) {
    a[i]     = ar[kb + i];
    a[8 + i] = ar[16 + kb + i];
  }

  // ---- B fragments: lane n holds column n, K striped over elements ----
  int ncol = lane & 15;
  int kb2  = (lane < 16) ? 0 : 8;
  v16h b0, b1;
#pragma unroll
  for (int i = 0; i < 16; ++i) {
    b0[i] = Bs[(kb2 + (i & 7) + ((i >> 3) << 4)) * HID + ncol];
    b1[i] = Bs[(kb2 + (i & 7) + ((i >> 3) << 4)) * HID + 16 + ncol];
  }

  v8f c = {};
  v8f d0 = __builtin_amdgcn_wmma_f32_16x16x32_f16(false, a, false, b0,
                                                  (short)0, c, false, false);
  v8f d1 = __builtin_amdgcn_wmma_f32_16x16x32_f16(false, a, false, b1,
                                                  (short)0, c, false, false);

  // ---- store D: VGPR v, lanes<16 -> M=v, lanes>=16 -> M=8+v, N=lane&15 ----
  int mb = (lane < 16) ? 0 : 8;
#pragma unroll
  for (int v = 0; v < 8; ++v) {
    float* hr = H + (size_t)(row0 + mb + v) * HID;
    hr[ncol]      = d0[v];
    hr[16 + ncol] = d1[v];
  }
}

// ---------------------------------------------------------------------------
// per-node attention scalars + init of m / denom / agg
// ---------------------------------------------------------------------------
__global__ void node_scalars(const float* __restrict__ h,
                             const float* __restrict__ asrc,
                             const float* __restrict__ adst,
                             float* __restrict__ ssrc, float* __restrict__ sdst,
                             float* __restrict__ m, float* __restrict__ denom,
                             float* __restrict__ agg, int N) {
  int n = blockIdx.x * blockDim.x + threadIdx.x;
  if (n >= N) return;
  const float* hr = h + (size_t)n * HID;
  float s1 = 0.0f, s2 = 0.0f;
#pragma unroll
  for (int c = 0; c < HID; ++c) {
    float v = hr[c];
    s1 += v * asrc[c];
    s2 += v * adst[c];
  }
  ssrc[n] = s1;
  sdst[n] = s2;
  m[n] = -INFINITY;
  denom[n] = 0.0f;
  float* ag = agg + (size_t)n * HID;
#pragma unroll
  for (int c = 0; c < HID; ++c) ag[c] = 0.0f;
}

// ---------------------------------------------------------------------------
// edge score + leaky relu + segment max (atomic)
// ---------------------------------------------------------------------------
__global__ void edge_score(const int* __restrict__ src, const int* __restrict__ dst,
                           const float* __restrict__ ea,   // [E,4]
                           const float* __restrict__ c4,
                           const float* __restrict__ ssrc,
                           const float* __restrict__ sdst,
                           float* __restrict__ score, float* __restrict__ m,
                           int E) {
  int e = blockIdx.x * blockDim.x + threadIdx.x;
  if (e >= E) return;
  int s = src[e], d = dst[e];
  const float* er = ea + (size_t)e * 4;
  float sc = ssrc[s] + sdst[d]
           + er[0] * c4[0] + er[1] * c4[1] + er[2] * c4[2] + er[3] * c4[3];
  sc = (sc > 0.0f) ? sc : NEG_SLOPE * sc;
  score[e] = sc;
  atomicMaxFloat(&m[d], sc);
}

// ---------------------------------------------------------------------------
// edge exp + segment sum (atomic); score overwritten in place with exp weight
// ---------------------------------------------------------------------------
__global__ void edge_exp(const int* __restrict__ dst,
                         float* __restrict__ score,
                         const float* __restrict__ m,
                         float* __restrict__ denom, int E) {
  int e = blockIdx.x * blockDim.x + threadIdx.x;
  if (e >= E) return;
  int d = dst[e];
  float w = __expf(score[e] - m[d]);
  score[e] = w;
  atomicAdd(&denom[d], w);
}

// ---------------------------------------------------------------------------
// weighted aggregation: agg[dst] += h[src] * alpha ; 8 threads per edge
// ---------------------------------------------------------------------------
__global__ void aggregate(const int* __restrict__ src, const int* __restrict__ dst,
                          const float* __restrict__ score,
                          const float* __restrict__ denom,
                          const float* __restrict__ h,
                          float* __restrict__ agg, int E) {
  long long t = (long long)blockIdx.x * blockDim.x + threadIdx.x;
  if (t >= (long long)E * 8) return;
  int e = (int)(t >> 3);
  int part = ((int)t & 7) * 4;
  int d = dst[e], s = src[e];
  float alpha = score[e] / (denom[d] + 1e-16f);
  const float4 hv = *(const float4*)(h + (size_t)s * HID + part);
  float* ap = agg + (size_t)d * HID + part;
  atomicAdd(ap + 0, hv.x * alpha);
  atomicAdd(ap + 1, hv.y * alpha);
  atomicAdd(ap + 2, hv.z * alpha);
  atomicAdd(ap + 3, hv.w * alpha);
}

// ---------------------------------------------------------------------------
// BN stats: per-channel sum & sumsq of (agg + bias) over nodes
// block = 256 = 8 node-rows x 32 channels; LDS reduce, one atomic per channel
// ---------------------------------------------------------------------------
__global__ __launch_bounds__(256)
void bn_stats(const float* __restrict__ agg, const float* __restrict__ bias,
              float* __restrict__ gsum, float* __restrict__ gsq, int N) {
  __shared__ float ssum[256], ssq[256];
  int tid = threadIdx.x;
  int c = tid & 31, r = tid >> 5;
  float b = bias[c];
  float acc = 0.0f, acc2 = 0.0f;
  for (int n = blockIdx.x * 8 + r; n < N; n += gridDim.x * 8) {
    float t = agg[(size_t)n * HID + c] + b;
    acc += t;
    acc2 += t * t;
  }
  ssum[tid] = acc;
  ssq[tid] = acc2;
  __syncthreads();
  if (r == 0) {
#pragma unroll
    for (int j = 1; j < 8; ++j) {
      acc  += ssum[c + 32 * j];
      acc2 += ssq[c + 32 * j];
    }
    atomicAdd(&gsum[c], acc);
    atomicAdd(&gsq[c], acc2);
  }
}

// ---------------------------------------------------------------------------
// BN apply + ReLU; writes f32 node embeddings and/or f16 next-layer input
// ---------------------------------------------------------------------------
__global__ void bn_apply(const float* __restrict__ agg,
                         const float* __restrict__ bias,
                         const float* __restrict__ gamma,
                         const float* __restrict__ beta,
                         const float* __restrict__ gsum,
                         const float* __restrict__ gsq,
                         float* __restrict__ outF32,      // nullable
                         _Float16* __restrict__ outF16,   // nullable
                         int N) {
  int idx = blockIdx.x * blockDim.x + threadIdx.x;
  if (idx >= N * HID) return;
  int c = idx & 31;
  float invN = 1.0f / (float)N;
  float mu = gsum[c] * invN;
  float var = gsq[c] * invN - mu * mu;
  float t = agg[idx] + bias[c];
  float y = (t - mu) * rsqrtf(var + BN_EPS) * gamma[c] + beta[c];
  y = fmaxf(y, 0.0f);
  if (outF32) outF32[idx] = y;
  if (outF16) outF16[idx] = (_Float16)y;
}

// ---------------------------------------------------------------------------
// graph pooling: graph_emb[batch[n]] += node_embs[n]
// ---------------------------------------------------------------------------
__global__ void graph_pool(const float* __restrict__ nodes,
                           const int* __restrict__ batch,
                           float* __restrict__ gemb, int N) {
  int idx = blockIdx.x * blockDim.x + threadIdx.x;
  if (idx >= N * HID) return;
  int n = idx >> 5, c = idx & 31;
  atomicAdd(&gemb[(size_t)batch[n] * HID + c], nodes[idx]);
}

// ---------------------------------------------------------------------------
// final FC: out[g,k] = gemb[g] . fcW[:,k] + fcb[k]
// ---------------------------------------------------------------------------
__global__ void fc_out(const float* __restrict__ gemb,
                       const float* __restrict__ fcW,  // [32,2]
                       const float* __restrict__ fcb,  // [2]
                       float* __restrict__ out, int G, int NC) {
  int idx = blockIdx.x * blockDim.x + threadIdx.x;
  if (idx >= G * NC) return;
  int g = idx / NC, k = idx % NC;
  float s = fcb[k];
  const float* gr = gemb + (size_t)g * HID;
#pragma unroll
  for (int c = 0; c < HID; ++c) s += gr[c] * fcW[c * NC + k];
  out[idx] = s;
}

// ---------------------------------------------------------------------------
// launch
// ---------------------------------------------------------------------------
static inline size_t alignup(size_t v) { return (v + 255) & ~(size_t)255; }

extern "C" void kernel_launch(void* const* d_in, const int* in_sizes, int n_in,
                              void* d_out, int out_size, void* d_ws, size_t ws_size,
                              hipStream_t stream) {
  (void)n_in; (void)ws_size;
  const float* x         = (const float*)d_in[0];
  const int*   edge_idx  = (const int*)d_in[1];
  const int*   batch     = (const int*)d_in[2];
  const float* edge_attr = (const float*)d_in[3];
  const float* W0        = (const float*)d_in[4];
  const float* W12       = (const float*)d_in[5];
  const float* att_src   = (const float*)d_in[6];
  const float* att_dst   = (const float*)d_in[7];
  const float* lin_edge  = (const float*)d_in[8];
  const float* att_edge  = (const float*)d_in[9];
  const float* bias      = (const float*)d_in[10];
  const float* bn_gamma  = (const float*)d_in[11];
  const float* bn_beta   = (const float*)d_in[12];
  const float* fc_W      = (const float*)d_in[13];
  const float* fc_b      = (const float*)d_in[14];

  const int IN_DIM = 7;
  const int NC = 2;
  const int N = in_sizes[0] / IN_DIM;          // 100000
  const int E = in_sizes[1] / 2;               // 2000000
  const int G = (out_size - N * HID) / (NC + HID);  // 2000

  const int* src = edge_idx;
  const int* dst = edge_idx + E;

  // d_out layout: out [G*NC] | node_embs [N*HID] | graph_emb [G*HID]
  float* out_logits = (float*)d_out;
  float* node_embs  = out_logits + (size_t)G * NC;
  float* graph_emb  = node_embs + (size_t)N * HID;

  // --- workspace carve ---
  char* w = (char*)d_ws;
  size_t off = 0;
  _Float16* x16 = (_Float16*)(w + off); off += alignup((size_t)N * HID * 2);
  _Float16* Wp  = (_Float16*)(w + off); off += alignup(HID * HID * 2);
  float* h      = (float*)(w + off);    off += alignup((size_t)N * HID * 4);
  float* ssrc   = (float*)(w + off);    off += alignup((size_t)N * 4);
  float* sdst   = (float*)(w + off);    off += alignup((size_t)N * 4);
  float* mmax   = (float*)(w + off);    off += alignup((size_t)N * 4);
  float* denom  = (float*)(w + off);    off += alignup((size_t)N * 4);
  float* agg    = (float*)(w + off);    off += alignup((size_t)N * HID * 4);
  float* score  = (float*)(w + off);    off += alignup((size_t)E * 4);
  float* c4     = (float*)(w + off);    off += alignup(4 * 4);
  float* bnsum  = (float*)(w + off);    off += alignup(HID * 4);
  float* bnsq   = (float*)(w + off);    off += alignup(HID * 4);

  const int B = 256;
  const int nRowTiles = (N + 15) / 16;

  // layer-0 input: padded f16 copy of x
  prep_x<<<(N + B - 1) / B, B, 0, stream>>>(x, x16, N, IN_DIM);

  for (int l = 0; l < 3; ++l) {
    const float* Wsrc = (l == 0) ? W0 : (W12 + (size_t)(l - 1) * HID * HID);
    int K = (l == 0) ? IN_DIM : HID;
    prep_W<<<1, 256, 0, stream>>>(Wsrc, K, Wp);
    init_layer<<<1, 64, 0, stream>>>(lin_edge + (size_t)l * 4 * HID,
                                     att_edge + (size_t)l * HID,
                                     c4, bnsum, bnsq);
    gemm_h<<<(nRowTiles + 7) / 8, 256, 0, stream>>>(x16, Wp, h, nRowTiles);
    node_scalars<<<(N + B - 1) / B, B, 0, stream>>>(
        h, att_src + (size_t)l * HID, att_dst + (size_t)l * HID,
        ssrc, sdst, mmax, denom, agg, N);
    edge_score<<<(E + B - 1) / B, B, 0, stream>>>(
        src, dst, edge_attr, c4, ssrc, sdst, score, mmax, E);
    edge_exp<<<(E + B - 1) / B, B, 0, stream>>>(dst, score, mmax, denom, E);
    {
      long long tot = (long long)E * 8;
      aggregate<<<(unsigned)((tot + B - 1) / B), B, 0, stream>>>(
          src, dst, score, denom, h, agg, E);
    }
    bn_stats<<<200, 256, 0, stream>>>(agg, bias + (size_t)l * HID, bnsum, bnsq, N);
    {
      float* f32out = (l == 2) ? node_embs : nullptr;
      _Float16* f16out = (l == 2) ? nullptr : x16;
      bn_apply<<<(N * HID + B - 1) / B, B, 0, stream>>>(
          agg, bias + (size_t)l * HID, bn_gamma + (size_t)l * HID,
          bn_beta + (size_t)l * HID, bnsum, bnsq, f32out, f16out, N);
    }
  }

  zero_f32<<<(G * HID + B - 1) / B, B, 0, stream>>>(graph_emb, G * HID);
  graph_pool<<<(N * HID + B - 1) / B, B, 0, stream>>>(node_embs, batch, graph_emb, N);
  fc_out<<<(G * NC + B - 1) / B, B, 0, stream>>>(graph_emb, fc_W, fc_b,
                                                 out_logits, G, NC);
}
